// Encoder_83305185673524
// MI455X (gfx1250) — compile-verified, matrix operand order checked by reading, not run on hardware
//
#include <hip/hip_runtime.h>

// LSTM encoder for MI455X (gfx1250, wave32, WMMA + async-to-LDS).
// Phase 1 (parallel): gather emb[tokens] -> bf16, init h exchange buffer.
// Phase 2 (persistent, 16 WGs): per step t, each WG computes its 128 gate
// columns of z = [x_t | h] @ [Wx;Wh] + b via v_wmma_f32_16x16x32_bf16 with
// the 768x128 bf16 weight slice resident in LDS (192KB). A-operand staging
// uses global_load_async_to_lds_b128 (ASYNCcnt). h is published through L2
// with an atomic-counter grid barrier.

#define E_DIM 256
#define H_DIM 512
#define B_DIM 64
#define T_DIM 1024
#define NG    2048          // 4*H
#define NWG   16
#define NCOL  128           // gate columns per WG
#define KTOT  768           // E + H
#define KT    24            // KTOT/32
#define NT    8             // NCOL/16

typedef __attribute__((ext_vector_type(16))) __bf16 v16bf;
typedef __attribute__((ext_vector_type(8)))  float  v8f;

union Frag {
    uint4 q[2];
    v16bf v;
};

__device__ inline unsigned short f2bf(float f) {
    unsigned u = __float_as_uint(f);
    unsigned r = ((u >> 16) & 1u) + 0x7FFFu;   // round-to-nearest-even
    return (unsigned short)((u + r) >> 16);
}
__device__ inline float sigm(float x) { return 1.0f / (1.0f + __expf(-x)); }
__device__ inline float tanh_f(float x) {
    x = fminf(fmaxf(x, -15.0f), 15.0f);
    float e = __expf(2.0f * x);
    return (e - 1.0f) / (e + 1.0f);
}

// ---------------- Phase 1: embedding gather + bf16 convert ----------------
__global__ void lstm_init_kernel(const int* __restrict__ tokens,
                                 const float* __restrict__ emb,
                                 const float* __restrict__ h0,
                                 unsigned short* __restrict__ x_bf,   // (T,B,E) bf16
                                 unsigned short* __restrict__ h_buf,  // 2 x (B,H) bf16
                                 unsigned* __restrict__ counter) {
    int row = blockIdx.x;            // t*64 + b
    int t = row >> 6, b = row & 63;
    int tok = tokens[b * T_DIM + t];
    const float* er = emb + (long)tok * E_DIM;
    int e = threadIdx.x;
    x_bf[(long)row * E_DIM + e] = f2bf(er[e]);
    if (row < B_DIM) {               // rows 0..63 init h_buf[0] (row == b)
        h_buf[row * H_DIM + e]       = f2bf(h0[row * H_DIM + e]);
        h_buf[row * H_DIM + e + 256] = f2bf(h0[row * H_DIM + e + 256]);
    }
    if (row == 0 && e == 0) *counter = 0u;
}

// ---------------- Phase 2: persistent recurrent kernel ----------------
__global__ void __launch_bounds__(256, 1)
lstm_rec_kernel(const int* __restrict__ tokens,
                const float* __restrict__ h0,
                const float* __restrict__ c0,
                const float* __restrict__ Wx,
                const float* __restrict__ Wh,
                const float* __restrict__ bias,
                const unsigned short* __restrict__ x_bf,
                unsigned short* __restrict__ h_buf,
                unsigned* __restrict__ counter,
                float* __restrict__ out) {
    extern __shared__ char smem[];
    unsigned short* Wl = (unsigned short*)smem;                   // 8*24 B-fragments, 192KB
    unsigned short* Al = (unsigned short*)(smem + 196608);        // A stage 64x768 bf16, 96KB
    float*          Zl = (float*)(smem + 196608);                 // z overlay 64x128 f32, 32KB
    float*          Bl = (float*)(smem + 196608 + 98304);         // bias slice, 512B

    const int tid  = threadIdx.x;
    const int wg   = blockIdx.x;
    const int lane = tid & 31;
    const int wave = tid >> 5;

    // --- stage weight slice into LDS, pre-swizzled into WMMA B-fragments ---
    // B-frag layout: lane L -> n = L&15, K half = (L>=16? 8:0); elem e -> K + (e<8? e : e+8)
    for (int i = tid; i < NT * KT * 32 * 16; i += 256) {
        int e  = i & 15;
        int ln = (i >> 4) & 31;
        int f  = i >> 9;                 // fragment id 0..191
        int kt = f % KT;
        int nt = f / KT;
        int k  = kt * 32 + ((ln >> 4) << 3) + (e < 8 ? e : e + 8);
        int n_local = nt * 16 + (ln & 15);
        int col = (n_local >> 5) * H_DIM + wg * 32 + (n_local & 31);
        float w = (k < E_DIM) ? Wx[(long)k * NG + col]
                              : Wh[(long)(k - E_DIM) * NG + col];
        Wl[i] = f2bf(w);
    }
    if (tid < NCOL) {
        int col = (tid >> 5) * H_DIM + wg * 32 + (tid & 31);
        Bl[tid] = bias[col];
    }

    // LDS byte offset of the A staging region (for async-to-LDS)
    const unsigned al_base = (unsigned)(size_t)(void*)Al;

    // --- per-thread recurrent state: thread owns (b, 8 hidden units) ---
    const int b  = tid & 63;
    const int jb = (tid >> 6) << 3;
    float hreg[8], creg[8];
#pragma unroll
    for (int jj = 0; jj < 8; jj++) {
        int col = wg * 32 + jb + jj;
        hreg[jj] = h0[b * H_DIM + col];
        creg[jj] = c0[b * H_DIM + col];
    }
    __syncthreads();

    unsigned target = 0;
    for (int t = 0; t < T_DIM; t++) {
        const unsigned short* hb_cur = h_buf + (t & 1) * (B_DIM * H_DIM);
        unsigned short*       hb_nxt = h_buf + ((t + 1) & 1) * (B_DIM * H_DIM);

        // --- stage A = [x_t | h_prev] (64 x 768 bf16) into LDS, async ---
        // 6144 b128 transfers, 24 per thread; each lane issues its own
        // (global addr, LDS offset) pair; tracked by ASYNCcnt.
        const uint4* xrow = (const uint4*)(x_bf + (long)t * B_DIM * E_DIM);
        const uint4* hrow = (const uint4*)hb_cur;
        for (int u = tid; u < 64 * 96; u += 256) {
            int r = u / 96, c = u % 96;
            const void* gp = (c < 32) ? (const void*)(xrow + r * 32 + c)
                                      : (const void*)(hrow + r * 64 + (c - 32));
            unsigned lp = al_base + (unsigned)(u * 16);
            asm volatile("global_load_async_to_lds_b128 %0, %1, off"
                         :: "v"(lp), "v"(gp) : "memory");
        }
        // prefetch next step's x slab into caches while we wait/compute
        if (t + 1 < T_DIM) {
            const char* nx = (const char*)(x_bf + (long)(t + 1) * B_DIM * E_DIM);
            __builtin_prefetch(nx + tid * 128, 0, 1);
        }
        asm volatile("s_wait_asynccnt 0x0" ::: "memory");
        __syncthreads();

        // --- GEMM: wave owns n-tile = wave; 4 m-tiles; K loop of 24 ---
        v8f acc[4] = {};
        const int khalf = ((lane >> 4) & 1) << 3;
        const int mrow  = lane & 15;
        for (int kt = 0; kt < KT; kt++) {
            Frag bf_;
            const uint4* bp = (const uint4*)(Wl + ((wave * KT + kt) * 32 + lane) * 16);
            bf_.q[0] = bp[0];
            bf_.q[1] = bp[1];
            int kb = kt * 32 + khalf;
#pragma unroll
            for (int mt = 0; mt < 4; mt++) {
                Frag af;
                const unsigned short* ap = Al + (mt * 16 + mrow) * KTOT + kb;
                af.q[0] = *(const uint4*)ap;
                af.q[1] = *(const uint4*)(ap + 16);
                acc[mt] = __builtin_amdgcn_wmma_f32_16x16x32_bf16(
                    false, af.v, false, bf_.v, (short)0, acc[mt], false, false);
            }
        }
        __syncthreads();   // all waves done reading Al before z overlay

        // --- scatter accumulators to z LDS (64 x 128 f32) ---
        {
            int n_local = wave * 16 + (lane & 15);
            int mo = (lane >> 4) << 3;    // 0 or 8
#pragma unroll
            for (int mt = 0; mt < 4; mt++)
#pragma unroll
                for (int r = 0; r < 8; r++)
                    Zl[(mt * 16 + mo + r) * NCOL + n_local] = acc[mt][r];
        }
        __syncthreads();

        // --- pointwise LSTM update for this thread's 8 hidden units ---
        int  tok = tokens[b * T_DIM + t];
        bool m   = (tok != 0);
#pragma unroll
        for (int jj = 0; jj < 8; jj++) {
            int jl = jb + jj;
            float zi = Zl[b * NCOL + jl]      + Bl[jl];
            float zf = Zl[b * NCOL + 32 + jl] + Bl[32 + jl];
            float zg = Zl[b * NCOL + 64 + jl] + Bl[64 + jl];
            float zo = Zl[b * NCOL + 96 + jl] + Bl[96 + jl];
            float ii = sigm(zi), ff = sigm(zf);
            float gg = tanh_f(zg), oo = sigm(zo);
            float cn = ff * creg[jj] + ii * gg;
            float hn = oo * tanh_f(cn);
            if (m) { creg[jj] = cn; hreg[jj] = hn; }
            int col = wg * 32 + jl;
            out[((long)b * T_DIM + t) * H_DIM + col] = hreg[jj];
            hb_nxt[b * H_DIM + col] = f2bf(hreg[jj]);
        }

        // --- grid barrier: publish h to all 16 WGs ---
        __threadfence();
        __syncthreads();
        target += NWG;
        if (tid == 0) {
            __hip_atomic_fetch_add(counter, 1u, __ATOMIC_ACQ_REL, __HIP_MEMORY_SCOPE_AGENT);
            while (__hip_atomic_load(counter, __ATOMIC_ACQUIRE, __HIP_MEMORY_SCOPE_AGENT) < target)
                __builtin_amdgcn_s_sleep(2);
        }
        __syncthreads();
    }

    // --- final h_T, c_T ---
    const long OUTS = (long)B_DIM * T_DIM * H_DIM;
#pragma unroll
    for (int jj = 0; jj < 8; jj++) {
        int col = wg * 32 + jb + jj;
        out[OUTS + b * H_DIM + col]                   = hreg[jj];
        out[OUTS + B_DIM * H_DIM + b * H_DIM + col]   = creg[jj];
    }
}

// ---------------- host launcher ----------------
extern "C" void kernel_launch(void* const* d_in, const int* in_sizes, int n_in,
                              void* d_out, int out_size, void* d_ws, size_t ws_size,
                              hipStream_t stream) {
    const int*   tokens = (const int*)d_in[0];
    const float* h0     = (const float*)d_in[1];
    const float* c0     = (const float*)d_in[2];
    const float* emb    = (const float*)d_in[3];
    const float* Wx     = (const float*)d_in[4];
    const float* Wh     = (const float*)d_in[5];
    const float* bias   = (const float*)d_in[6];
    float*       out    = (float*)d_out;

    char* ws = (char*)d_ws;
    unsigned*       counter = (unsigned*)ws;
    unsigned short* x_bf    = (unsigned short*)(ws + 256);
    unsigned short* h_buf   = (unsigned short*)(ws + 256 + (size_t)B_DIM * T_DIM * E_DIM * 2);

    lstm_init_kernel<<<B_DIM * T_DIM, 256, 0, stream>>>(tokens, emb, h0, x_bf, h_buf, counter);

    size_t lds = 196608 + 98304 + 512;   // weights + A/z overlay + bias = ~295KB
    lstm_rec_kernel<<<NWG, 256, lds, stream>>>(tokens, h0, c0, Wx, Wh, bias,
                                               x_bf, h_buf, counter, out);
}
